// MultiHeadAttention_67877663146521
// MI455X (gfx1250) — matrix operand order repaired and checked by reading, not compiled, not run on hardware
//
#include <hip/hip_runtime.h>

typedef __attribute__((ext_vector_type(16))) __bf16 v16bf;
typedef __attribute__((ext_vector_type(8)))  __bf16 v8bf;
typedef __attribute__((ext_vector_type(8)))  float  v8f;

#define D_MODEL 768
#define SEQ     2048
#define BATCH   4
#define HEADS   12
#define HDIM    64
#define ROWS    (BATCH*SEQ)   // 8192

// ---------------------------------------------------------------------------
// WMMA fragment loaders (wave32; M/N/C layouts per CDNA5 ISA §7.12.2; K order is this code's)
//
// A (16x32 bf16, M x K): lane L -> row M = L%16 ; half-slot i -> K =
//   16*(L/16) + i   -> one contiguous 16-elem K run (same order as B).
// B (32x16 bf16, K x N): lane L -> col N = L%16 ; half-slot i ->
//   K = 16*(L/16) + i                 -> one contiguous 16-elem K run.
// C/D (16x16 f32): VGPR r, lane L -> M = r + 8*(L/16), N = L%16.
// ---------------------------------------------------------------------------

__device__ inline v16bf load_a_frag(const __bf16* __restrict__ base, int stride,
                                    int m0, int kb, int lane) {
  const int m = m0 + (lane & 15);
  const int g = lane >> 4;
  const __bf16* p = base + (size_t)m * stride + kb + 16 * g;
  v8bf lo = *(const v8bf*)p;          // K = kb+16g .. +7
  v8bf hi = *(const v8bf*)(p + 8);    // K = kb+16g+8 .. +15
  v16bf a;
#pragma unroll
  for (int i = 0; i < 8; i++) { a[i] = lo[i]; a[i + 8] = hi[i]; }
  return a;
}

// B fragment from "transposed" storage: element (k,n) lives at base[n*stride + k]
__device__ inline v16bf load_b_fragT(const __bf16* __restrict__ base, int stride,
                                     int n0, int kb, int lane) {
  const int n = n0 + (lane & 15);
  const int g = lane >> 4;
  return *(const v16bf*)(base + (size_t)n * stride + kb + 16 * g);
}

__device__ inline v8f wmma_bf16(v16bf a, v16bf b, v8f c) {
  return __builtin_amdgcn_wmma_f32_16x16x32_bf16(false, a, false, b,
                                                 (short)0, c, false, false);
}

__device__ inline float hmax16(float v) {
#pragma unroll
  for (int m = 1; m < 16; m <<= 1) v = fmaxf(v, __shfl_xor(v, m, 32));
  return v;
}
__device__ inline float hsum16(float v) {
#pragma unroll
  for (int m = 1; m < 16; m <<= 1) v += __shfl_xor(v, m, 32);
  return v;
}

// ---------------------------------------------------------------------------
// Kernel 0: dtype conversion + weight transpose (bf16 [N][K])
// ---------------------------------------------------------------------------
__global__ void __launch_bounds__(256, 1)
prep_kernel(const float* __restrict__ X,
            const float* __restrict__ Wq, const float* __restrict__ Wk,
            const float* __restrict__ Wv, const float* __restrict__ Wo,
            __bf16* __restrict__ Xb,
            __bf16* __restrict__ WqT, __bf16* __restrict__ WkT,
            __bf16* __restrict__ WvT, __bf16* __restrict__ WoT) {
  const size_t tid = (size_t)blockIdx.x * blockDim.x + threadIdx.x;
  const size_t stride = (size_t)gridDim.x * blockDim.x;
  for (size_t j = tid; j < (size_t)ROWS * D_MODEL; j += stride)
    Xb[j] = (__bf16)X[j];
  for (size_t j = tid; j < (size_t)D_MODEL * D_MODEL; j += stride) {
    const int k = (int)(j / D_MODEL), n = (int)(j % D_MODEL);
    const size_t o = (size_t)n * D_MODEL + k;
    WqT[o] = (__bf16)Wq[j];
    WkT[o] = (__bf16)Wk[j];
    WvT[o] = (__bf16)Wv[j];
    WoT[o] = (__bf16)Wo[j];
  }
}

// ---------------------------------------------------------------------------
// Kernel 1: QKV projection.  One wave -> 16(M) x 64(N) tile of X @ W.
// grid = (512 m-tiles, 9), block = 128 (4 waves); job = by*4+wave in [0,36):
//   mat = job/12 (Q,K,V), ngroup = job%12.
// Q,K stored [b,h,n,64] bf16; V stored transposed [b,h,64,n] bf16.
// ---------------------------------------------------------------------------
__global__ void __launch_bounds__(128, 1)
qkv_kernel(const __bf16* __restrict__ Xb,
           const __bf16* __restrict__ WqT, const __bf16* __restrict__ WkT,
           const __bf16* __restrict__ WvT,
           __bf16* __restrict__ Q, __bf16* __restrict__ K,
           __bf16* __restrict__ Vt) {
  const int lane = threadIdx.x & 31;
  const int wave = threadIdx.x >> 5;
  const int g = lane >> 4;
  const int job = blockIdx.y * 4 + wave;   // 0..35
  const int mat = job / 12;                // 0=Q 1=K 2=V
  const int n0 = (job % 12) * 64;
  const int m0 = blockIdx.x * 16;

  const __bf16* WT = (mat == 0) ? WqT : (mat == 1) ? WkT : WvT;

  v8f c[4] = {};
  for (int kb = 0; kb < D_MODEL; kb += 32) {
    v16bf a = load_a_frag(Xb, D_MODEL, m0, kb, lane);
#pragma unroll
    for (int t = 0; t < 4; t++) {
      v16bf b = load_b_fragT(WT, D_MODEL, n0 + 16 * t, kb, lane);
      c[t] = wmma_bf16(a, b, c[t]);
    }
  }

#pragma unroll
  for (int t = 0; t < 4; t++) {
#pragma unroll
    for (int r = 0; r < 8; r++) {
      const int row = m0 + r + 8 * g;          // global token row
      const int bidx = row >> 11, tok = row & (SEQ - 1);
      const int n = n0 + 16 * t + (lane & 15);
      const int h = n >> 6, dim = n & 63;
      const __bf16 v = (__bf16)c[t][r];
      if (mat == 2)
        Vt[(((size_t)bidx * HEADS + h) * HDIM + dim) * SEQ + tok] = v;
      else if (mat == 1)
        K[(((size_t)bidx * HEADS + h) * SEQ + tok) * HDIM + dim] = v;
      else
        Q[(((size_t)bidx * HEADS + h) * SEQ + tok) * HDIM + dim] = v;
    }
  }
}

// ---------------------------------------------------------------------------
// Kernel 2: causal flash attention.  One wave per 16-row q-tile, k-step 32.
// grid = (32, 48 bh), block = 128 (4 waves), 4 KB LDS for C->A re-layout of P.
// ---------------------------------------------------------------------------
__global__ void __launch_bounds__(128, 1)
attn_kernel(const __bf16* __restrict__ Q, const __bf16* __restrict__ K,
            const __bf16* __restrict__ Vt, __bf16* __restrict__ ctx) {
  __shared__ __bf16 smem[4][16 * 32];

  const int lane = threadIdx.x & 31;
  const int wave = threadIdx.x >> 5;
  const int g = lane >> 4;
  const int bh = blockIdx.y;                 // 0..47
  const int b = bh / HEADS, h = bh % HEADS;
  const int qt = blockIdx.x * 4 + wave;      // 0..127
  const int qbase = qt * 16;

  const __bf16* Qh = Q + (size_t)bh * SEQ * HDIM;
  const __bf16* Kh = K + (size_t)bh * SEQ * HDIM;
  const __bf16* Vh = Vt + (size_t)bh * HDIM * SEQ;
  __bf16* sm = smem[wave];

  // Q tile 16x64 lives in registers for the whole loop (2 A-fragments)
  const v16bf aq0 = load_a_frag(Qh, HDIM, qbase, 0, lane);
  const v16bf aq1 = load_a_frag(Qh, HDIM, qbase, 32, lane);

  v8f o[4] = {};
  float mrow[8], lrow[8];
#pragma unroll
  for (int r = 0; r < 8; r++) { mrow[r] = -__builtin_inff(); lrow[r] = 0.f; }

  // causal: need tokens 0 .. qbase+15  ->  kt < qbase+16 (kt_last <= 2016)
  for (int kt = 0; kt < qbase + 16; kt += 32) {
    // S = Q @ K^T for 32 tokens : two 16x16 C tiles, contraction over 64 dims
    v8f s[2] = {};
#pragma unroll
    for (int t = 0; t < 2; t++) {
      const int tok = kt + 16 * t + (lane & 15);           // always < SEQ
      const __bf16* kp = Kh + (size_t)tok * HDIM + 16 * g;
      v16bf b0 = *(const v16bf*)kp;                        // dims 16g..16g+15
      v16bf b1 = *(const v16bf*)(kp + 32);                 // dims 32+16g..
      s[t] = wmma_bf16(aq0, b0, s[t]);
      s[t] = wmma_bf16(aq1, b1, s[t]);
    }

    // scale + causal mask + row max (rows live across 16 lanes of a half)
    float pmax[8];
#pragma unroll
    for (int r = 0; r < 8; r++) {
      const int qrow = qbase + r + 8 * g;
      const float v0 = (kt + (lane & 15) <= qrow) ? s[0][r] * 0.125f
                                                  : -__builtin_inff();
      const float v1 = (kt + 16 + (lane & 15) <= qrow) ? s[1][r] * 0.125f
                                                       : -__builtin_inff();
      s[0][r] = v0; s[1][r] = v1;
      pmax[r] = hmax16(fmaxf(v0, v1));
    }

    // online softmax update; stage P (bf16) in LDS for C->A re-layout
#pragma unroll
    for (int r = 0; r < 8; r++) {
      const float mnew = fmaxf(mrow[r], pmax[r]);
      const float scale = __expf(mrow[r] - mnew);
      const float p0 = __expf(s[0][r] - mnew);
      const float p1 = __expf(s[1][r] - mnew);
      sm[(r + 8 * g) * 32 + (lane & 15)] = (__bf16)p0;
      sm[(r + 8 * g) * 32 + 16 + (lane & 15)] = (__bf16)p1;
      lrow[r] = lrow[r] * scale + hsum16(p0 + p1);
      mrow[r] = mnew;
      o[0][r] *= scale; o[1][r] *= scale; o[2][r] *= scale; o[3][r] *= scale;
    }

    // same-wave LDS RAW: order stores before the A-fragment gather
    asm volatile("s_wait_dscnt 0" ::: "memory");

    const v16bf pa = load_a_frag(sm, 32, 0, 0, lane);

    // O += P(16x32) @ V(32 x 64)   (V stored dim-major -> contiguous tokens)
#pragma unroll
    for (int d = 0; d < 4; d++) {
      const __bf16* vp =
          Vh + (size_t)(16 * d + (lane & 15)) * SEQ + kt + 16 * g;
      v16bf bv = *(const v16bf*)vp;
      o[d] = wmma_bf16(pa, bv, o[d]);
    }
  }

  // normalize and store ctx as [b, token, 768] bf16
#pragma unroll
  for (int r = 0; r < 8; r++) lrow[r] = 1.0f / lrow[r];
#pragma unroll
  for (int d = 0; d < 4; d++)
#pragma unroll
    for (int r = 0; r < 8; r++) {
      const int row = qbase + r + 8 * g;
      const int n = h * HDIM + 16 * d + (lane & 15);
      ctx[((size_t)b * SEQ + row) * D_MODEL + n] = (__bf16)(o[d][r] * lrow[r]);
    }
}

// ---------------------------------------------------------------------------
// Kernel 3: output projection  out = ctx @ Wo + bo   (fp32 out)
// grid = (512, 3), block = 128 (4 waves); wave -> 16x64 tile.
// ---------------------------------------------------------------------------
__global__ void __launch_bounds__(128, 1)
proj_kernel(const __bf16* __restrict__ ctx, const __bf16* __restrict__ WoT,
            const float* __restrict__ bo, float* __restrict__ out) {
  const int lane = threadIdx.x & 31;
  const int wave = threadIdx.x >> 5;
  const int g = lane >> 4;
  const int n0 = (blockIdx.y * 4 + wave) * 64;   // 0..704
  const int m0 = blockIdx.x * 16;

  v8f c[4] = {};
  for (int kb = 0; kb < D_MODEL; kb += 32) {
    v16bf a = load_a_frag(ctx, D_MODEL, m0, kb, lane);
#pragma unroll
    for (int t = 0; t < 4; t++) {
      v16bf b = load_b_fragT(WoT, D_MODEL, n0 + 16 * t, kb, lane);
      c[t] = wmma_bf16(a, b, c[t]);
    }
  }

#pragma unroll
  for (int t = 0; t < 4; t++) {
    const int n = n0 + 16 * t + (lane & 15);
    const float bias = bo[n];
#pragma unroll
    for (int r = 0; r < 8; r++) {
      const int row = m0 + r + 8 * g;
      out[(size_t)row * D_MODEL + n] = c[t][r] + bias;
    }
  }
}

// ---------------------------------------------------------------------------
extern "C" void kernel_launch(void* const* d_in, const int* in_sizes, int n_in,
                              void* d_out, int out_size, void* d_ws,
                              size_t ws_size, hipStream_t stream) {
  const float* X  = (const float*)d_in[0];
  const float* Wq = (const float*)d_in[1];
  const float* Wk = (const float*)d_in[2];
  const float* Wv = (const float*)d_in[3];
  const float* Wo = (const float*)d_in[4];
  const float* bo = (const float*)d_in[5];

  // workspace carve-up (~67.6 MB, all 4KB-aligned offsets)
  char* ws = (char*)d_ws;
  const size_t XB = (size_t)ROWS * D_MODEL * sizeof(__bf16);      // 12.58 MB
  const size_t WT = (size_t)D_MODEL * D_MODEL * sizeof(__bf16);   // 1.18 MB
  __bf16* Xb  = (__bf16*)ws; ws += XB;
  __bf16* WqT = (__bf16*)ws; ws += WT;
  __bf16* WkT = (__bf16*)ws; ws += WT;
  __bf16* WvT = (__bf16*)ws; ws += WT;
  __bf16* WoT = (__bf16*)ws; ws += WT;
  __bf16* Qb  = (__bf16*)ws; ws += XB;
  __bf16* Kb  = (__bf16*)ws; ws += XB;
  __bf16* Vtb = (__bf16*)ws; ws += XB;
  __bf16* ctx = (__bf16*)ws; ws += XB;

  prep_kernel<<<dim3(4096), dim3(256), 0, stream>>>(X, Wq, Wk, Wv, Wo, Xb,
                                                    WqT, WkT, WvT, WoT);
  qkv_kernel<<<dim3(512, 9), dim3(128), 0, stream>>>(Xb, WqT, WkT, WvT, Qb,
                                                     Kb, Vtb);
  attn_kernel<<<dim3(32, 48), dim3(128), 0, stream>>>(Qb, Kb, Vtb, ctx);
  proj_kernel<<<dim3(512, 3), dim3(128), 0, stream>>>(ctx, WoT, bo,
                                                      (float*)d_out);
}
